// MyModule_18279380812077
// MI455X (gfx1250) — compile-verified, matrix operand order checked by reading, not run on hardware
//
#include <hip/hip_runtime.h>
#include <hip/hip_bf16.h>

typedef __attribute__((ext_vector_type(16))) _Float16 v16h;
typedef __attribute__((ext_vector_type(8)))  _Float16 v8h;
typedef __attribute__((ext_vector_type(8)))  float    v8f;
typedef __attribute__((ext_vector_type(4)))  float    v4f;
typedef __attribute__((ext_vector_type(4)))  unsigned int u32x4;
typedef __attribute__((ext_vector_type(4)))  int          i32x4;
typedef __attribute__((ext_vector_type(8)))  int          i32x8;

#ifndef __has_builtin
#define __has_builtin(x) 0
#endif

#if defined(__AMDGCN__) && __has_builtin(__builtin_amdgcn_tensor_load_to_lds)
#define HAVE_TDM 1
#else
#define HAVE_TDM 0
#endif

// amdgpu-toolchain (therock-10.0 headers) ships the TDM header and uses the
// 6-arg builtin; ROCm 7.2 uses the 5-arg form.
#if __has_include(<hip/amd_detail/amd_gfx1250_TDM.h>)
#define TDM_SIX_ARGS 1
#else
#define TDM_SIX_ARGS 0
#endif

namespace {
constexpr int kN = 262144;   // rows
constexpr int kK = 512;      // in_dim
constexpr int kO = 128;      // out_dim
constexpr int kGemmBlocks = kN / 128;   // 2048 blocks, 128 rows each
}

// ---------------------------------------------------------------------------
// W [128,512] fp32 -> f16 copy in workspace (one-time, tiny)
// ---------------------------------------------------------------------------
__global__ __launch_bounds__(256) void wcvt_kernel(const float* __restrict__ W,
                                                   _Float16* __restrict__ Wh) {
  int i = (blockIdx.x * 256 + threadIdx.x) * 4;
  v4f w = *(const v4f*)(W + i);
#pragma unroll
  for (int j = 0; j < 4; ++j) Wh[i + j] = (_Float16)w[j];
}

// ---------------------------------------------------------------------------
// GEMM: out = x @ W^T + b  via v_wmma_f32_16x16x32_f16, fp32 accumulate.
// Block = 256 threads = 8 waves; wave w owns rows [blk*128 + w*16, +16).
// W(f16) is staged into LDS once per block (TDM tensor_load_to_lds when the
// builtin exists, else a plain cooperative copy); B fragments then come from
// ds_load_b128 so no global-load wait sits in front of each WMMA.
// A (x) fragments are double-buffered: next K-step's loads are issued before
// the current 8 WMMAs so the streaming HBM traffic overlaps the matrix ops.
// ---------------------------------------------------------------------------
__global__ __launch_bounds__(256) void gemm_kernel(const float* __restrict__ x,
                                                   const _Float16* __restrict__ Wh,
                                                   const float* __restrict__ b,
                                                   float* __restrict__ out,
                                                   float* __restrict__ partials) {
  __shared__ _Float16 Wlds[kO * kK];     // 128 KB: full f16 W, row-major [128][512]
  __shared__ float red[256];

  const int lane  = threadIdx.x & 31;
  const int wave  = threadIdx.x >> 5;                 // 0..7
  const int rowBase = blockIdx.x * 128 + wave * 16;
  const int mrow  = lane & 15;                        // row (A) / col (B) within tile
  const int khalf = (lane >> 4) * 8;                  // 0 for lanes 0-15, 8 for 16-31
  const int half8 = (lane >> 4) * 8;                  // C/D row offset for this lane half

  // ---- stage W(f16) into LDS --------------------------------------------
#if HAVE_TDM
  if (threadIdx.x == 0) {
    typedef __attribute__((address_space(3))) _Float16 lds_f16;
    unsigned lds_addr = (unsigned)(size_t)(lds_f16*)&Wlds[0];
    unsigned long long ga = (unsigned long long)(const void*)Wh;

    // D# group0: count=1 | lds_addr | global_addr[56:0] | type=2
    u32x4 g0;
    g0[0] = 1u;
    g0[1] = lds_addr;
    g0[2] = (unsigned)(ga & 0xFFFFFFFFull);
    g0[3] = (unsigned)((ga >> 32) & 0x01FFFFFFull) | (2u << 30);

    // D# group1: data_size=2B; tensor 512x128; tile 512x128; dim0_stride=512
    i32x8 g1;
    g1[0] = 0x00010000;                                  // data_size=1 (2 bytes)
    g1[1] = (int)((unsigned)(kK & 0xFFFF) << 16);        // tensor_dim0 lo16
    g1[2] = (int)((unsigned)(kK >> 16) |
                  ((unsigned)(kO & 0xFFFF) << 16));      // dim0 hi16 | dim1 lo16
    g1[3] = (int)((unsigned)(kO >> 16) |
                  ((unsigned)(kK & 0xFFFF) << 16));      // dim1 hi16 | tile_dim0
    g1[4] = kO;                                          // tile_dim1=128, tile_dim2=0
    g1[5] = kK;                                          // tensor_dim0_stride lo32
    g1[6] = 0;
    g1[7] = 0;

    i32x4 z4 = {0, 0, 0, 0};
#if TDM_SIX_ARGS
    i32x8 z8 = {0, 0, 0, 0, 0, 0, 0, 0};
    __builtin_amdgcn_tensor_load_to_lds(g0, g1, z4, z4, z8, 0);
#else
    __builtin_amdgcn_tensor_load_to_lds(g0, g1, z4, z4, 0);
#endif
    __builtin_amdgcn_s_wait_tensorcnt(0);
  }
  __syncthreads();
#else
  for (int i = threadIdx.x * 8; i < kO * kK; i += 256 * 8)
    *(v8h*)(Wlds + i) = *(const v8h*)(Wh + i);
  __syncthreads();
#endif

  const float* __restrict__ xrow = x + (size_t)(rowBase + mrow) * kK;

  v8f acc[8];
#pragma unroll
  for (int ct = 0; ct < 8; ++ct) acc[ct] = (v8f){};

  // ---- prime A double-buffer (K-step 0) ----------------------------------
  const float* p0 = xrow + khalf;
  v4f a0 = *(const v4f*)(p0);
  v4f a1 = *(const v4f*)(p0 + 4);
  v4f a2 = *(const v4f*)(p0 + 16);
  v4f a3 = *(const v4f*)(p0 + 20);

  for (int kk = 0; kk < 16; ++kk) {
    // issue next K-step's x loads early (kk==15 reloads kk 0; harmless)
    const float* pn = xrow + ((kk + 1) & 15) * 32 + khalf;
    v4f n0 = *(const v4f*)(pn);
    v4f n1 = *(const v4f*)(pn + 4);
    v4f n2 = *(const v4f*)(pn + 16);
    v4f n3 = *(const v4f*)(pn + 20);

    // A fragment (16x32 f16): lanes 0-15 hold K 0-7 & 16-23, lanes 16-31 K 8-15 & 24-31
    v16h afrag;
#pragma unroll
    for (int j = 0; j < 4; ++j) {
      afrag[j]      = (_Float16)a0[j];
      afrag[4 + j]  = (_Float16)a1[j];
      afrag[8 + j]  = (_Float16)a2[j];
      afrag[12 + j] = (_Float16)a3[j];
    }

#pragma unroll
    for (int ct = 0; ct < 8; ++ct) {
      // B fragment (32x16 f16), B[k][n] = W[n][k]; symmetric layout to A; from LDS
      const _Float16* q = Wlds + (ct * 16 + mrow) * kK + kk * 32 + khalf;
      v8h b0 = *(const v8h*)(q);
      v8h b1 = *(const v8h*)(q + 16);
      v16h bfrag;
#pragma unroll
      for (int j = 0; j < 8; ++j) { bfrag[j] = b0[j]; bfrag[8 + j] = b1[j]; }

      acc[ct] = __builtin_amdgcn_wmma_f32_16x16x32_f16(
          false, afrag, false, bfrag, (short)0, acc[ct], false, false);
    }

    a0 = n0; a1 = n1; a2 = n2; a3 = n3;
  }

  // ---- bias add, store, and per-lane sum for threshold
  float lsum = 0.0f;
#pragma unroll
  for (int ct = 0; ct < 8; ++ct) {
    float bv = b[ct * 16 + mrow];
#pragma unroll
    for (int r = 0; r < 8; ++r) {
      float v = acc[ct][r] + bv;            // lane half gives M = r (+8), N = mrow
      out[(size_t)(rowBase + half8 + r) * kO + ct * 16 + mrow] = v;
      lsum += v;
    }
  }

  // deterministic block reduction of this block's 128x128 tile sum
  red[threadIdx.x] = lsum;
  __syncthreads();
#pragma unroll
  for (int d = 128; d > 0; d >>= 1) {
    if (threadIdx.x < d) red[threadIdx.x] += red[threadIdx.x + d];
    __syncthreads();
  }
  if (threadIdx.x == 0) partials[blockIdx.x] = red[0];
}

// ---------------------------------------------------------------------------
// threshold = (sum of all out elements) / N   (== mean of per-row sums)
// ---------------------------------------------------------------------------
__global__ __launch_bounds__(256) void threshold_kernel(const float* __restrict__ partials,
                                                        float* __restrict__ thresh) {
  __shared__ float s[256];
  float sum = 0.0f;
#pragma unroll
  for (int i = 0; i < 8; ++i) sum += partials[threadIdx.x * 8 + i];
  s[threadIdx.x] = sum;
  __syncthreads();
#pragma unroll
  for (int d = 128; d > 0; d >>= 1) {
    if (threadIdx.x < d) s[threadIdx.x] += s[threadIdx.x + d];
    __syncthreads();
  }
  if (threadIdx.x == 0) thresh[0] = s[0] / (float)kN;
}

// ---------------------------------------------------------------------------
// Compaction: fill -1, per-block counts, single-block scan, ordered scatter
// ---------------------------------------------------------------------------
__global__ __launch_bounds__(256) void fill_kernel(int* __restrict__ idx) {
  idx[blockIdx.x * 256 + threadIdx.x] = -1;
}

__global__ __launch_bounds__(256) void count_kernel(const float* __restrict__ mask,
                                                    const float* __restrict__ thresh,
                                                    int* __restrict__ counts) {
  __shared__ int s[256];
  const float th = thresh[0];
  const int base = blockIdx.x * 1024 + threadIdx.x * 4;
  int c = 0;
#pragma unroll
  for (int j = 0; j < 4; ++j) c += (mask[base + j] > th) ? 1 : 0;
  s[threadIdx.x] = c;
  __syncthreads();
#pragma unroll
  for (int d = 128; d > 0; d >>= 1) {
    if (threadIdx.x < d) s[threadIdx.x] += s[threadIdx.x + d];
    __syncthreads();
  }
  if (threadIdx.x == 0) counts[blockIdx.x] = s[0];
}

__global__ __launch_bounds__(256) void scan_kernel(const int* __restrict__ counts,
                                                   int* __restrict__ offsets) {
  __shared__ int s[256];
  const int t = threadIdx.x;
  const int orig = counts[t];
  s[t] = orig;
  __syncthreads();
  for (int d = 1; d < 256; d <<= 1) {
    int add = (t >= d) ? s[t - d] : 0;
    __syncthreads();
    s[t] += add;
    __syncthreads();
  }
  offsets[t] = s[t] - orig;   // exclusive prefix
}

__global__ __launch_bounds__(256) void scatter_kernel(const float* __restrict__ mask,
                                                      const float* __restrict__ thresh,
                                                      const int* __restrict__ offsets,
                                                      int* __restrict__ idx) {
  __shared__ int s[256];
  const float th = thresh[0];
  const int t = threadIdx.x;
  const int base = blockIdx.x * 1024 + t * 4;
  int pred[4];
  int c = 0;
#pragma unroll
  for (int j = 0; j < 4; ++j) { pred[j] = (mask[base + j] > th) ? 1 : 0; c += pred[j]; }
  const int orig = c;
  s[t] = c;
  __syncthreads();
  for (int d = 1; d < 256; d <<= 1) {
    int add = (t >= d) ? s[t - d] : 0;
    __syncthreads();
    s[t] += add;
    __syncthreads();
  }
  int pos = offsets[blockIdx.x] + s[t] - orig;
#pragma unroll
  for (int j = 0; j < 4; ++j) {
    if (pred[j]) idx[pos++] = base + j;
  }
}

// ---------------------------------------------------------------------------
extern "C" void kernel_launch(void* const* d_in, const int* in_sizes, int n_in,
                              void* d_out, int out_size, void* d_ws, size_t ws_size,
                              hipStream_t stream) {
  (void)in_sizes; (void)n_in; (void)out_size; (void)ws_size;

  const float* x    = (const float*)d_in[0];   // [N, 512]
  const float* mask = (const float*)d_in[1];   // [N]
  const float* W    = (const float*)d_in[2];   // [128, 512]
  const float* b    = (const float*)d_in[3];   // [128]

  float* out = (float*)d_out;                         // [N, 128] fp32
  int*   idx = (int*)(out + (size_t)kN * kO);         // [N] int32 (padded -1)

  char* ws = (char*)d_ws;
  float*    partials = (float*)(ws + 0);       // 2048 floats
  float*    thresh   = (float*)(ws + 8192);    // 1 float
  int*      counts   = (int*)(ws + 8448);      // 256 ints
  int*      offsets  = (int*)(ws + 9472);      // 256 ints
  _Float16* Wh       = (_Float16*)(ws + 16384);// 65536 halves (128 KB)

  wcvt_kernel<<<64, 256, 0, stream>>>(W, Wh);
  gemm_kernel<<<kGemmBlocks, 256, 0, stream>>>(x, Wh, b, out, partials);
  threshold_kernel<<<1, 256, 0, stream>>>(partials, thresh);
  fill_kernel<<<kN / 256, 256, 0, stream>>>(idx);
  count_kernel<<<256, 256, 0, stream>>>(mask, thresh, counts);
  scan_kernel<<<1, 256, 0, stream>>>(counts, offsets);
  scatter_kernel<<<256, 256, 0, stream>>>(mask, thresh, offsets, idx);
}
